// CrossAttentionFusion_2087354106618
// MI455X (gfx1250) — matrix operand order repaired and checked by reading, not compiled, not run on hardware
//
#include <hip/hip_runtime.h>

typedef __attribute__((ext_vector_type(4)))  int    i4;
typedef __attribute__((ext_vector_type(8)))  float  v8f;
typedef __attribute__((ext_vector_type(16))) __bf16 v16bf;

union BF16Tile { i4 q[2]; v16bf v; };

#define NB    8
#define NCH   64
#define NPIX  4096
#define MT    64
#define NWAVE 8

static __device__ __forceinline__ unsigned short f2bf(float f) {
  union { float f; unsigned u; } x; x.f = f;
  unsigned r = x.u + 0x7fffu + ((x.u >> 16) & 1u);   // round-to-nearest-even
  return (unsigned short)(r >> 16);
}

// ---- CDNA5 async global->LDS copy (ASYNCcnt-tracked, bypasses VGPRs) ------
static __device__ __forceinline__ void async_b128(unsigned lds_off, unsigned voff,
                                                  const void* sbase) {
  asm volatile("global_load_async_to_lds_b128 %0, %1, %2"
               :: "v"(lds_off), "v"(voff), "s"(sbase) : "memory");
}
static __device__ __forceinline__ void wait_async0() {
  asm volatile("s_wait_asynccnt 0" ::: "memory");
}
static __device__ __forceinline__ unsigned lds_addr(const void* p) {
  return (unsigned)(unsigned long long)p;   // low 32 bits of generic ptr = LDS offset
}

// ---------------------------------------------------------------------------
// Phase 1a (VALU, only 0.07 GF): q -> bf16 [B][N][32] (K-padded), k -> [B][8][M]
// ---------------------------------------------------------------------------
__global__ __launch_bounds__(256) void qk_proj(
    const float* __restrict__ xq, const float* __restrict__ xkv,
    const float* __restrict__ Wq, const float* __restrict__ bq,
    const float* __restrict__ Wk, const float* __restrict__ bk,
    unsigned short* __restrict__ qb, unsigned short* __restrict__ kb)
{
  __shared__ float WqL[8 * 64], WkL[8 * 64], bqL[8], bkL[8];
  const int tid = threadIdx.x;
  for (int i = tid; i < 512; i += 256) { WqL[i] = Wq[i]; WkL[i] = Wk[i]; }
  if (tid < 8) { bqL[tid] = bq[tid]; bkL[tid] = bk[tid]; }
  __syncthreads();

  const int g = blockIdx.x * 256 + tid;   // b*N + n
  const int b = g >> 12, n = g & (NPIX - 1);
  {
    const float* xp = xq + (size_t)b * NCH * NPIX + n;
    float acc[8] = {};
    for (int c = 0; c < 64; ++c) {
      float x = xp[(size_t)c * NPIX];
#pragma unroll
      for (int d = 0; d < 8; ++d) acc[d] += WqL[d * 64 + c] * x;
    }
    unsigned* qp = (unsigned*)(qb + (size_t)g * 32);
#pragma unroll
    for (int d = 0; d < 4; ++d)
      qp[d] = (unsigned)f2bf(acc[2 * d] + bqL[2 * d]) |
              ((unsigned)f2bf(acc[2 * d + 1] + bqL[2 * d + 1]) << 16);
#pragma unroll
    for (int d = 4; d < 16; ++d) qp[d] = 0u;
  }
  {
    const float* xp = xkv + (size_t)b * NCH * NPIX + n;
    float acc[8] = {};
    for (int c = 0; c < 64; ++c) {
      float x = xp[(size_t)c * NPIX];
#pragma unroll
      for (int d = 0; d < 8; ++d) acc[d] += WkL[d * 64 + c] * x;
    }
#pragma unroll
    for (int d = 0; d < 8; ++d)
      kb[((size_t)b * 8 + d) * NPIX + n] = f2bf(acc[d] + bkL[d]);
  }
}

// ---------------------------------------------------------------------------
// Phase 1b (WMMA): v[b][m][e] = Wv*xkv + bv.   64x64 x (64 x 128-col strips).
// ---------------------------------------------------------------------------
__global__ __launch_bounds__(256) void v_gemm(
    const float* __restrict__ xkv, const float* __restrict__ Wv,
    const float* __restrict__ bv, unsigned short* __restrict__ vb)
{
  __shared__ __align__(16) unsigned short Wlds[64][64];   // bf16 W[f][c]
  __shared__ __align__(16) unsigned short Xlds[64][128];  // bf16 X[c][m]
  __shared__ __align__(16) unsigned short Ylds[128][64];  // bf16 Y[m][f]
  __shared__ float bvl[64];
  const int tid = threadIdx.x, wave = tid >> 5, lane = tid & 31;
  const int half = lane >> 4, ln = lane & 15;
  const int b = blockIdx.x >> 5, m0 = (blockIdx.x & 31) * 128;

  for (int i = tid; i < 4096; i += 256) ((unsigned short*)Wlds)[i] = f2bf(Wv[i]);
  if (tid < 64) bvl[tid] = bv[tid];
  {
    int c = tid >> 2, co = (tid & 3) * 32;
    const float* xp = xkv + ((size_t)b * 64 + c) * NPIX + m0 + co;
#pragma unroll
    for (int j = 0; j < 32; ++j) Xlds[c][co + j] = f2bf(xp[j]);
  }
  __syncthreads();

  const int f0 = (wave & 3) * 16, mb = (wave >> 2) * 64;
  BF16Tile aw[2];
#pragma unroll
  for (int ks = 0; ks < 2; ++ks) {
    const unsigned short* wp = &Wlds[f0 + ln][32 * ks + half * 8];
    aw[ks].q[0] = *(const i4*)(wp);
    aw[ks].q[1] = *(const i4*)(wp + 16);
  }
#pragma unroll
  for (int t = 0; t < 4; ++t) {
    v8f acc = 0.0f;
#pragma unroll
    for (int ks = 0; ks < 2; ++ks) {
      BF16Tile bx;
#pragma unroll
      for (int i = 0; i < 16; ++i)
        ((unsigned short*)&bx)[i] = Xlds[32 * ks + 8 * half + (i & 7) + 16 * (i >> 3)][mb + 16 * t + ln];
      acc = __builtin_amdgcn_wmma_f32_16x16x32_bf16(
          false, aw[ks].v, false, bx.v, (short)0, acc, false, false);
    }
#pragma unroll
    for (int r = 0; r < 8; ++r) {
      int f = f0 + r + 8 * half;
      Ylds[mb + 16 * t + ln][f] = f2bf(acc[r] + bvl[f]);
    }
  }
  __syncthreads();
  {
    int m = tid >> 1, co = (tid & 1) * 32;
    unsigned short* op = vb + ((size_t)b * NPIX + m0 + m) * 64 + co;
    const unsigned short* yp = &Ylds[m][co];
    *(i4*)(op)      = *(const i4*)(yp);
    *(i4*)(op + 8)  = *(const i4*)(yp + 8);
    *(i4*)(op + 16) = *(const i4*)(yp + 16);
    *(i4*)(op + 24) = *(const i4*)(yp + 24);
  }
}

// ---------------------------------------------------------------------------
// Phase 2: flash attention with double-buffered async-LDS K/V staging.
// ---------------------------------------------------------------------------
__global__ __launch_bounds__(256) void attn(
    const unsigned short* __restrict__ qb,   // [B][N][32] bf16
    const unsigned short* __restrict__ kb,   // [B][8][M]  bf16
    const unsigned short* __restrict__ vb,   // [B][M][64] bf16
    float* __restrict__ Obuf)                // [B][N][64] f32
{
  __shared__ __align__(16) unsigned short Klds[2][32][64];   // rows 8..15, 24..31 stay 0; 16..23 unset
  __shared__ __align__(16) unsigned short Vlds[2][64][64];
  __shared__ __align__(16) unsigned short Plds[NWAVE][16][64];

  const int tid  = threadIdx.x;
  const int wave = tid >> 5, lane = tid & 31;
  const int half = lane >> 4, ln = lane & 15;
  const int b    = blockIdx.x >> 5;
  const int n0   = ((blockIdx.x & 31) * NWAVE + wave) * 16;

  const unsigned short* kbB = kb + (size_t)b * 8 * NPIX;
  const unsigned short* vbB = vb + (size_t)b * NPIX * 64;

  // zero K-pad rows 8..15 and 24..31 of both buffers once (the j mask skips 16..23)
  for (int i = tid; i < 2 * 24 * 32; i += 256) {
    int buf = i >= 24 * 32, j = i & (24 * 32 - 1);
    ((unsigned*)&Klds[buf][8][0])[j] = 0u;
  }

  // persistent Q A-operand
  BF16Tile aq;
  {
    const unsigned short* qp =
        qb + ((size_t)b * NPIX + (n0 + ln)) * 32 + half * 8;
    aq.q[0] = *(const i4*)(qp);
    aq.q[1] = *(const i4*)(qp + 16);
  }

  // staging geometry (per thread)
  const int krow = tid >> 3, kco = (tid & 7) * 8;      // threads 0..63
  const int vrow = tid >> 2, vco = (tid & 3) * 16;     // all threads

  // issue chunk 0
  if (tid < 64)
    async_b128(lds_addr(&Klds[0][krow][kco]),
               (unsigned)((krow * NPIX + kco) * 2), kbB);
  {
    unsigned l = lds_addr(&Vlds[0][vrow][vco]);
    unsigned g = (unsigned)((vrow * 64 + vco) * 2);
    async_b128(l, g, vbB);
    async_b128(l + 16, g + 16, vbB);
  }
  wait_async0();
  __syncthreads();

  v8f O0, O1, O2, O3, mrow, lrow;
#pragma unroll
  for (int r = 0; r < 8; ++r) {
    O0[r] = O1[r] = O2[r] = O3[r] = 0.f;
    mrow[r] = -1e30f; lrow[r] = 0.f;
  }

  int cur = 0;
  for (int it = 0; it < NPIX / MT; ++it, cur ^= 1) {
    // prefetch next chunk into the other buffer (async, no VGPR traffic)
    const int m1 = (it + 1) * MT;
    if (m1 < NPIX) {
      if (tid < 64)
        async_b128(lds_addr(&Klds[cur ^ 1][krow][kco]),
                   (unsigned)((krow * NPIX + m1 + kco) * 2), kbB);
      unsigned l = lds_addr(&Vlds[cur ^ 1][vrow][vco]);
      unsigned g = (unsigned)(((m1 + vrow) * 64 + vco) * 2);
      async_b128(l, g, vbB);
      async_b128(l + 16, g + 16, vbB);
    }

    // ---- S = Q * K -------------------------------------------------------
    v8f S[4];
#pragma unroll
    for (int s = 0; s < 4; ++s) {
      BF16Tile bk;
#pragma unroll
      for (int i = 0; i < 16; ++i)
        ((unsigned short*)&bk)[i] = Klds[cur][8 * half + (i & 7) + 16 * (i >> 3)][16 * s + ln];
      v8f z = 0.0f;
      S[s] = __builtin_amdgcn_wmma_f32_16x16x32_bf16(
          false, aq.v, false, bk.v, (short)0, z, false, false);
    }

    // ---- online softmax --------------------------------------------------
    v8f mnew, scal;
#pragma unroll
    for (int r = 0; r < 8; ++r) {
      float mx = fmaxf(fmaxf(S[0][r], S[1][r]), fmaxf(S[2][r], S[3][r]));
      mx = fmaxf(mx, __shfl_xor(mx, 1));
      mx = fmaxf(mx, __shfl_xor(mx, 2));
      mx = fmaxf(mx, __shfl_xor(mx, 4));
      mx = fmaxf(mx, __shfl_xor(mx, 8));
      mnew[r] = fmaxf(mrow[r], mx);
      scal[r] = __expf(mrow[r] - mnew[r]);
      mrow[r] = mnew[r];
    }
#pragma unroll
    for (int r = 0; r < 8; ++r) {
      float p0 = __expf(S[0][r] - mnew[r]);
      float p1 = __expf(S[1][r] - mnew[r]);
      float p2 = __expf(S[2][r] - mnew[r]);
      float p3 = __expf(S[3][r] - mnew[r]);
      S[0][r] = p0; S[1][r] = p1; S[2][r] = p2; S[3][r] = p3;
      float rs = p0 + p1 + p2 + p3;
      rs += __shfl_xor(rs, 1);
      rs += __shfl_xor(rs, 2);
      rs += __shfl_xor(rs, 4);
      rs += __shfl_xor(rs, 8);
      lrow[r] = lrow[r] * scal[r] + rs;
      O0[r] *= scal[r]; O1[r] *= scal[r]; O2[r] *= scal[r]; O3[r] *= scal[r];
    }

    // ---- P layout swap via wave-private LDS (LDS is in-order per wave) ---
#pragma unroll
    for (int s = 0; s < 4; ++s)
#pragma unroll
      for (int r = 0; r < 8; ++r)
        Plds[wave][r + 8 * half][16 * s + ln] = f2bf(S[s][r]);
    asm volatile("" ::: "memory");   // keep ds_store before ds_load

    // ---- O += P * V ------------------------------------------------------
#pragma unroll
    for (int ks = 0; ks < 2; ++ks) {
      BF16Tile ap;
      const unsigned short* pp = &Plds[wave][ln][32 * ks + half * 8];
      ap.q[0] = *(const i4*)(pp);
      ap.q[1] = *(const i4*)(pp + 16);
#pragma unroll
      for (int t = 0; t < 4; ++t) {
        BF16Tile bvt;
#pragma unroll
        for (int i = 0; i < 16; ++i)
          ((unsigned short*)&bvt)[i] = Vlds[cur][32 * ks + 8 * half + (i & 7) + 16 * (i >> 3)][16 * t + ln];
        v8f& Ot = (t == 0 ? O0 : t == 1 ? O1 : t == 2 ? O2 : O3);
        Ot = __builtin_amdgcn_wmma_f32_16x16x32_bf16(
            false, ap.v, false, bvt.v, (short)0, Ot, false, false);
      }
    }

    wait_async0();       // next buffer staged (this wave's portion)
    __syncthreads();     // everyone done staging next + reading cur
  }

#pragma unroll
  for (int r = 0; r < 8; ++r) {
    float inv = 1.0f / lrow[r];
    int n = n0 + r + 8 * half;
    float* op = Obuf + ((size_t)b * NPIX + n) * 64;
    op[ 0 + ln] = O0[r] * inv;
    op[16 + ln] = O1[r] * inv;
    op[32 + ln] = O2[r] * inv;
    op[48 + ln] = O3[r] * inv;
  }
}

// ---------------------------------------------------------------------------
// Phase 3 (WMMA): out = gamma * (Wo*O + bo) + feat_query
// ---------------------------------------------------------------------------
__global__ __launch_bounds__(256) void o_gemm(
    const float* __restrict__ Obuf, const float* __restrict__ Wo,
    const float* __restrict__ bo, const float* __restrict__ gamma,
    const float* __restrict__ xq, float* __restrict__ out)
{
  __shared__ __align__(16) unsigned short Wlds[64][64];   // bf16 Wo[f][e]
  __shared__ __align__(16) unsigned short Xlds[64][128];  // bf16 O^T[e][n]
  __shared__ float bol[64];
  const int tid = threadIdx.x, wave = tid >> 5, lane = tid & 31;
  const int half = lane >> 4, ln = lane & 15;
  const int b = blockIdx.x >> 5, n0 = (blockIdx.x & 31) * 128;

  for (int i = tid; i < 4096; i += 256) ((unsigned short*)Wlds)[i] = f2bf(Wo[i]);
  if (tid < 64) bol[tid] = bo[tid];
  {
    int n = tid >> 1, eo = (tid & 1) * 32;   // transpose O[n][e] -> X[e][n]
    const float* op = Obuf + ((size_t)b * NPIX + n0 + n) * 64 + eo;
#pragma unroll
    for (int j = 0; j < 32; ++j) Xlds[eo + j][n] = f2bf(op[j]);
  }
  __syncthreads();

  const int f0 = (wave & 3) * 16, nb = (wave >> 2) * 64;
  BF16Tile aw[2];
#pragma unroll
  for (int ks = 0; ks < 2; ++ks) {
    const unsigned short* wp = &Wlds[f0 + ln][32 * ks + half * 8];
    aw[ks].q[0] = *(const i4*)(wp);
    aw[ks].q[1] = *(const i4*)(wp + 16);
  }
  const float gm = gamma[0];
#pragma unroll
  for (int t = 0; t < 4; ++t) {
    v8f acc = 0.0f;
#pragma unroll
    for (int ks = 0; ks < 2; ++ks) {
      BF16Tile bx;
#pragma unroll
      for (int i = 0; i < 16; ++i)
        ((unsigned short*)&bx)[i] = Xlds[32 * ks + 8 * half + (i & 7) + 16 * (i >> 3)][nb + 16 * t + ln];
      acc = __builtin_amdgcn_wmma_f32_16x16x32_bf16(
          false, aw[ks].v, false, bx.v, (short)0, acc, false, false);
    }
#pragma unroll
    for (int r = 0; r < 8; ++r) {
      int f = f0 + r + 8 * half;
      int n = n0 + nb + 16 * t + ln;
      size_t idx = ((size_t)b * 64 + f) * NPIX + n;
      out[idx] = gm * (acc[r] + bol[f]) + xq[idx];
    }
  }
}

// ---------------------------------------------------------------------------
extern "C" void kernel_launch(void* const* d_in, const int* in_sizes, int n_in,
                              void* d_out, int out_size, void* d_ws, size_t ws_size,
                              hipStream_t stream) {
  const float* xq    = (const float*)d_in[0];
  const float* xkv   = (const float*)d_in[1];
  const float* Wq    = (const float*)d_in[2];
  const float* bq    = (const float*)d_in[3];
  const float* Wk    = (const float*)d_in[4];
  const float* bk    = (const float*)d_in[5];
  const float* Wv    = (const float*)d_in[6];
  const float* bv    = (const float*)d_in[7];
  const float* Wo    = (const float*)d_in[8];
  const float* bo    = (const float*)d_in[9];
  const float* gamma = (const float*)d_in[10];
  float* out = (float*)d_out;

  char* ws = (char*)d_ws;
  unsigned short* qb = (unsigned short*)ws; ws += (size_t)NB * NPIX * 32 * 2; // 2 MB
  unsigned short* kb = (unsigned short*)ws; ws += (size_t)NB * 8 * NPIX * 2;  // 0.5 MB
  unsigned short* vb = (unsigned short*)ws; ws += (size_t)NB * NPIX * 64 * 2; // 4 MB
  float* Obuf        = (float*)ws;                                            // 8 MB

  qk_proj<<<NB * NPIX / 256, 256, 0, stream>>>(xq, xkv, Wq, bq, Wk, bk, qb, kb);
  v_gemm <<<NB * 32,          256, 0, stream>>>(xkv, Wv, bv, vb);
  attn   <<<NB * 32,          256, 0, stream>>>(qb, kb, vb, Obuf);
  o_gemm <<<NB * 32,          256, 0, stream>>>(Obuf, Wo, bo, gamma, xq, out);
}
